// DynamicsEncoder_1786706395434
// MI455X (gfx1250) — matrix operand order repaired and checked by reading, not compiled, not run on hardware
//
#include <hip/hip_runtime.h>

// Problem dims (fixed by the reference)
#define Bq   8
#define Sq   512
#define Eq   128          // encoding dim / E
#define Hq   512          // hidden
#define Mq   (Bq * Sq)    // 4096 rows
#define Oq   (Eq * Eq)    // 16384 = ENC*ENC

typedef __attribute__((ext_vector_type(16))) _Float16 v16h;
typedef __attribute__((ext_vector_type(8)))  float    v8f;

union V16 { uint4 u[2]; v16h v; };

__device__ __forceinline__ unsigned short f2h_bits(float f) {
  _Float16 h = (_Float16)f;
  return *(unsigned short*)&h;
}

// ---------------- fp32 -> fp16 conversion ----------------
__global__ void k_convert(const float* __restrict__ src,
                          unsigned short* __restrict__ dst, int n) {
  int i = blockIdx.x * blockDim.x + threadIdx.x;
  if (i < n) dst[i] = f2h_bits(src[i]);
}

// A fragment: 16x32 f16, A row-major [*, K], rows mbase..mbase+15, k = kbase..kbase+31
// Lane L: m = mbase + (L&15), h = L>>4; elems 0..7 -> k = kbase + h*8 + j;
// elems 8..15 -> k = kbase + 16 + h*8 + (j-8).  Two contiguous 16B loads.
__device__ __forceinline__ v16h load_frag_A(const unsigned short* __restrict__ A,
                                            int K, int mbase, int kbase, int lane) {
  const int h = lane >> 4, r = lane & 15;
  const unsigned short* p = A + (size_t)(mbase + r) * K + kbase + h * 8;
  V16 t;
  t.u[0] = *(const uint4*)(p);
  t.u[1] = *(const uint4*)(p + 16);
  return t.v;
}

// B fragment: 32x16 f16, sourced from Bt row-major [N, K] (Bt row n == B column n).
// Lane L: n = nbase + (L&15), h = L>>4; elems 0..7 -> k = kbase + h*8 + j;
// elems 8..15 -> k = kbase + 16 + h*8 + (j-8).
__device__ __forceinline__ v16h load_frag_B(const unsigned short* __restrict__ Bt,
                                            int K, int nbase, int kbase, int lane) {
  const int h = lane >> 4, c = lane & 15;
  const unsigned short* p = Bt + (size_t)(nbase + c) * K + kbase + h * 8;
  V16 t;
  t.u[0] = *(const uint4*)(p);
  t.u[1] = *(const uint4*)(p + 16);
  return t.v;
}

// ---------------- Stage 1: hidden = tanh(A @ W1^T + b1) ----------------
// M=4096, N=512, K=128. One 16x16 tile per wave, 4 k-steps of WMMA f16 K=32.
__global__ void k_hidden(const unsigned short* __restrict__ Ah,   // [4096,128] f16
                         const unsigned short* __restrict__ W1h,  // [512,128]  f16 (Bt form)
                         const float* __restrict__ b1,
                         unsigned short* __restrict__ hid) {      // [4096,512] f16
  const int lane = threadIdx.x & 31;
  const int w    = blockIdx.x * (blockDim.x >> 5) + (threadIdx.x >> 5);
  const int NT   = Hq / 16;            // 32
  const int mt = w / NT, nt = w % NT;
  const int mbase = mt * 16, nbase = nt * 16;

  v8f c = {};
  #pragma unroll
  for (int kb = 0; kb < Eq; kb += 32) {
    v16h a = load_frag_A(Ah,  Eq, mbase, kb, lane);
    v16h b = load_frag_B(W1h, Eq, nbase, kb, lane);
    c = __builtin_amdgcn_wmma_f32_16x16x32_f16(false, a, false, b, (short)0, c, false, false);
  }

  const int h = lane >> 4;
  const int n = nbase + (lane & 15);
  const float bias = b1[n];
  #pragma unroll
  for (int r = 0; r < 8; ++r) {
    float v = tanhf(c[r] + bias);
    hid[(size_t)(mbase + r + 8 * h) * Hq + n] = f2h_bits(v);
  }
}

// ---------------- Stage 2: Wa = hidden @ W2^T + b2 ----------------
// M=4096, N=16384, K=512. Each wave computes a 32x64 tile: 2 A fragments x
// 4 B fragments -> 8 WMMAs per k-step with 12 b128 loads (1.5 loads/WMMA),
// 16 k-steps. fp32 output to workspace.
__global__ void k_wa(const unsigned short* __restrict__ hid,  // [4096,512]  f16
                     const unsigned short* __restrict__ W2h,  // [16384,512] f16 (Bt form)
                     const float* __restrict__ b2,
                     float* __restrict__ Wa) {                // [4096,16384] f32
  const int lane = threadIdx.x & 31;
  const int w    = blockIdx.x * (blockDim.x >> 5) + (threadIdx.x >> 5);
  const int NT4  = Oq / 64;            // 256 column-tiles of width 64
  const int mt = w / NT4, nt4 = w % NT4;
  const int mbase = mt * 32, nbase = nt4 * 64;

  v8f acc00 = {}, acc01 = {}, acc02 = {}, acc03 = {};
  v8f acc10 = {}, acc11 = {}, acc12 = {}, acc13 = {};
  for (int kb = 0; kb < Hq; kb += 32) {
    v16h a0 = load_frag_A(hid, Hq, mbase,      kb, lane);
    v16h a1 = load_frag_A(hid, Hq, mbase + 16, kb, lane);
    v16h b0 = load_frag_B(W2h, Hq, nbase +  0, kb, lane);
    v16h b1v = load_frag_B(W2h, Hq, nbase + 16, kb, lane);
    v16h b2v = load_frag_B(W2h, Hq, nbase + 32, kb, lane);
    v16h b3v = load_frag_B(W2h, Hq, nbase + 48, kb, lane);
    acc00 = __builtin_amdgcn_wmma_f32_16x16x32_f16(false, a0, false, b0,  (short)0, acc00, false, false);
    acc01 = __builtin_amdgcn_wmma_f32_16x16x32_f16(false, a0, false, b1v, (short)0, acc01, false, false);
    acc02 = __builtin_amdgcn_wmma_f32_16x16x32_f16(false, a0, false, b2v, (short)0, acc02, false, false);
    acc03 = __builtin_amdgcn_wmma_f32_16x16x32_f16(false, a0, false, b3v, (short)0, acc03, false, false);
    acc10 = __builtin_amdgcn_wmma_f32_16x16x32_f16(false, a1, false, b0,  (short)0, acc10, false, false);
    acc11 = __builtin_amdgcn_wmma_f32_16x16x32_f16(false, a1, false, b1v, (short)0, acc11, false, false);
    acc12 = __builtin_amdgcn_wmma_f32_16x16x32_f16(false, a1, false, b2v, (short)0, acc12, false, false);
    acc13 = __builtin_amdgcn_wmma_f32_16x16x32_f16(false, a1, false, b3v, (short)0, acc13, false, false);
  }

  const int h   = lane >> 4;
  const int col = lane & 15;
  v8f accs[2][4] = {{acc00, acc01, acc02, acc03}, {acc10, acc11, acc12, acc13}};
  #pragma unroll
  for (int m2 = 0; m2 < 2; ++m2) {
    #pragma unroll
    for (int i = 0; i < 4; ++i) {
      const int n = nbase + i * 16 + col;
      const float bias = b2[n];
      #pragma unroll
      for (int r = 0; r < 8; ++r) {
        Wa[(size_t)(mbase + m2 * 16 + r + 8 * h) * Oq + n] = accs[m2][i][r] + bias;
      }
    }
  }
}

// ---------------- Stage 3: sequential attractor scan ----------------
// One block per batch (8 blocks, 128 threads). et kept in LDS; each step
// does a coalesced [128]x[128x128] vec-mat, clamps to +-5, emits output.
// Prefetch next step's Wa slab to hide HBM latency across 511 serial steps.
__global__ void k_scan(const float* __restrict__ Wa,   // [4096,16384]
                       const float* __restrict__ e0,   // [128]
                       float* __restrict__ out) {      // [8,512,128]
  const int b = blockIdx.x;
  const int t = threadIdx.x;   // 0..127 -> output element e
  __shared__ float et[Eq];

  float e = e0[t];
  et[t] = e;
  out[((size_t)b * Sq + 0) * Eq + t] = e;
  __syncthreads();

  for (int s = 1; s < Sq; ++s) {
    const float* __restrict__ W = Wa + ((size_t)(b * Sq + s)) * Oq;
    float a0 = 0.f, a1 = 0.f, a2 = 0.f, a3 = 0.f;
    #pragma unroll 8
    for (int d = 0; d < Eq; d += 4) {
      a0 = fmaf(et[d + 0], W[(d + 0) * Eq + t], a0);
      a1 = fmaf(et[d + 1], W[(d + 1) * Eq + t], a1);
      a2 = fmaf(et[d + 2], W[(d + 2) * Eq + t], a2);
      a3 = fmaf(et[d + 3], W[(d + 3) * Eq + t], a3);
      if (s + 1 < Sq)  // warm next step's 64KB slab into cache
        __builtin_prefetch(W + Oq + d * Eq + t, 0, 3);
    }
    float v = (a0 + a1) + (a2 + a3);
    v = fminf(5.0f, fmaxf(-5.0f, v));
    __syncthreads();
    et[t] = v;
    __syncthreads();
    out[((size_t)b * Sq + s) * Eq + t] = v;
  }
}

// ---------------- host-side orchestration ----------------
extern "C" void kernel_launch(void* const* d_in, const int* in_sizes, int n_in,
                              void* d_out, int out_size, void* d_ws, size_t ws_size,
                              hipStream_t stream) {
  (void)in_sizes; (void)n_in; (void)out_size; (void)ws_size;

  const float* act = (const float*)d_in[0];  // [8,512,128]
  const float* W1  = (const float*)d_in[1];  // [512,128]
  const float* b1  = (const float*)d_in[2];  // [512]
  const float* W2  = (const float*)d_in[3];  // [16384,512]
  const float* b2  = (const float*)d_in[4];  // [16384]
  const float* e0  = (const float*)d_in[5];  // [128]
  float* out = (float*)d_out;

  // Workspace layout (all 256B-aligned by construction)
  char* ws = (char*)d_ws;
  const size_t nA  = (size_t)Mq * Eq;   // 524288
  const size_t nW1 = (size_t)Hq * Eq;   // 65536
  const size_t nW2 = (size_t)Oq * Hq;   // 8388608
  const size_t nH  = (size_t)Mq * Hq;   // 2097152
  unsigned short* Ah  = (unsigned short*)(ws);
  unsigned short* W1h = (unsigned short*)(ws + nA * 2);
  unsigned short* W2h = (unsigned short*)(ws + nA * 2 + nW1 * 2);
  unsigned short* hid = (unsigned short*)(ws + nA * 2 + nW1 * 2 + nW2 * 2);
  float*          Wa  = (float*)        (ws + nA * 2 + nW1 * 2 + nW2 * 2 + nH * 2);

  // fp32 -> fp16 operand conversion
  k_convert<<<(unsigned)((nA  + 255) / 256), 256, 0, stream>>>(act, Ah,  (int)nA);
  k_convert<<<(unsigned)((nW1 + 255) / 256), 256, 0, stream>>>(W1,  W1h, (int)nW1);
  k_convert<<<(unsigned)((nW2 + 255) / 256), 256, 0, stream>>>(W2,  W2h, (int)nW2);

  // Stage 1: 256 M-tiles x 32 N-tiles = 8192 waves / 8 waves-per-block
  k_hidden<<<1024, 256, 0, stream>>>(Ah, W1h, b1, hid);

  // Stage 2: 128 (M/32)-tiles x 256 (N/64)-tiles = 32768 waves / 8 per block
  k_wa<<<4096, 256, 0, stream>>>(hid, W2h, b2, Wa);

  // Stage 3: sequential scan, one block per batch
  k_scan<<<Bq, Eq, 0, stream>>>(Wa, e0, out);
}
